// BilstmCrfModel_56418690400395
// MI455X (gfx1250) — matrix operand order repaired and checked by reading, not compiled, not run on hardware
//
#include <hip/hip_runtime.h>
#include <hip/hip_bf16.h>

// ---------------- problem dims ----------------
#define SB   512      // sequence length S
#define BB   128      // batch B
#define EMBD 128      // embedding dim (= K of input GEMM)
#define HD   128      // hidden per direction (= K of recurrent GEMM)
#define G4   512      // 4*H gate width
#define LL   9        // number of tags

typedef __attribute__((ext_vector_type(16))) __bf16 v16bf;
typedef __attribute__((ext_vector_type(8)))  float  v8f;
typedef __attribute__((ext_vector_type(4)))  unsigned u32x4;
typedef __attribute__((ext_vector_type(8)))  int      i32x8;
typedef __attribute__((ext_vector_type(4)))  int      i32x4;

// ---------------- helpers ----------------
__device__ __forceinline__ __bf16 to_bf16(float f) {
    unsigned u = __builtin_bit_cast(unsigned, f);
    u += 0x7fffu + ((u >> 16) & 1u);            // round-to-nearest-even
    unsigned short s = (unsigned short)(u >> 16);
    return __builtin_bit_cast(__bf16, s);
}

__device__ __forceinline__ float sigmf(float x) {
    return 1.0f / (1.0f + __expf(-x));
}

// ---- CDNA5 async global->LDS (ASYNCcnt path) ----
// VGLOBAL async encoding: VDST = per-lane LDS byte address, VADDR = 64-bit global addr.
// LDS byte offset == low 32 bits of the flat shared pointer (aperture rule, ISA 10.2).
__device__ __forceinline__ void async_b128(void* ldsPtr, const void* gPtr) {
    unsigned lds_off = (unsigned)(uintptr_t)ldsPtr;
    unsigned long long ga = (unsigned long long)(uintptr_t)gPtr;
    asm volatile("global_load_async_to_lds_b128 %0, %1, off"
                 :: "v"(lds_off), "v"(ga) : "memory");
}

__device__ __forceinline__ void wait_async0() {
#if __has_builtin(__builtin_amdgcn_s_wait_asynccnt)
    __builtin_amdgcn_s_wait_asynccnt(0);
#else
    asm volatile("s_wait_asynccnt 0x0" ::: "memory");
#endif
}

// ---- CDNA5 TDM: DMA a 2D bf16 tile (rows x cols, row stride = cols) into LDS ----
// D# built per CDNA5 ISA. Issued wave-uniform.
__device__ __forceinline__ void tdm_load_2d_bf16(void* ldsPtr, const void* gPtr,
                                                 unsigned cols, unsigned rows) {
#if __has_builtin(__builtin_amdgcn_tensor_load_to_lds)
    unsigned long long ga = (unsigned long long)(uintptr_t)gPtr;
    u32x4 g0 = {};
    g0[0] = 1u;                                         // count=1, user descriptor
    g0[1] = (unsigned)(uintptr_t)ldsPtr;                // lds_addr (bytes)
    g0[2] = (unsigned)(ga & 0xffffffffu);               // global_addr[31:0]
    g0[3] = (unsigned)((ga >> 32) & 0x01ffffffu) | 0x80000000u;  // addr[56:32] | type=2
    i32x8 g1 = {};
    g1[0] = (int)(1u << 16);                            // data_size=1 (2 bytes), wg_mask=0
    g1[1] = (int)(cols << 16);                          // tensor_dim0[15:0]
    g1[2] = (int)(rows << 16);                          // tensor_dim0[31:16]=0, tensor_dim1[15:0]
    g1[3] = (int)(cols << 16);                          // tensor_dim1[31:16]=0, tile_dim0
    g1[4] = (int)rows;                                  // tile_dim1 (tile_dim2=0)
    g1[5] = (int)cols;                                  // tensor_dim0_stride[31:0]
    i32x4 g2 = {}, g3 = {};                             // dims 2/3 unused (zero)
#if __clang_major__ >= 23
    i32x8 g4 = {};
    __builtin_amdgcn_tensor_load_to_lds(g0, g1, g2, g3, g4, 0);
#else
    __builtin_amdgcn_tensor_load_to_lds(g0, g1, g2, g3, 0);
#endif
    __builtin_amdgcn_s_wait_tensorcnt(0);
#endif
}

union Frag32 { v16bf v; uint4 u[2]; };

// A-matrix 16x32 bf16 fragment (M rows in lds row-major, leading dim ldk).
// lane<16: m=m0+lane, koff=0 ; lane>=16: m=m0+lane-16, koff=8
// elements 0..7  -> K = kb+koff+{0..7} ; elements 8..15 -> K = kb+16+koff+{0..7}
__device__ __forceinline__ v16bf load_a_frag(const __bf16* lds, int m0, int kb, int ldk) {
    const int lane = threadIdx.x & 31;
    const int m    = m0 + (lane & 15);
    const int koff = (lane >> 4) << 3;
    const __bf16* p = lds + m * ldk + kb + koff;
    Frag32 f;
    f.u[0] = *(const uint4*)(p);
    f.u[1] = *(const uint4*)(p + 16);
    return f.v;
}

// B-matrix 32x16 bf16 fragment from row-major W[n][k] (leading dim ldk).
// lane<16: n=n0+lane, K=kb+{0..7},kb+16+{0..7} ; lane>=16: n=n0+lane-16, K=kb+8+{0..7},kb+24+{0..7}
__device__ __forceinline__ v16bf load_b_frag(const __bf16* lds, int n0, int kb, int ldk) {
    const int lane = threadIdx.x & 31;
    const int n    = n0 + (lane & 15);
    const int koff = (lane >> 4) << 3;
    const __bf16* p = lds + n * ldk + kb + koff;
    Frag32 f;
    f.u[0] = *(const uint4*)(p);
    f.u[1] = *(const uint4*)(p + 16);
    return f.v;
}

#define WMMA_BF16(a, b, c) \
    __builtin_amdgcn_wmma_f32_16x16x32_bf16(false, (a), false, (b), (short)0, (c), false, false)

// ---------------- K0: cast weights to bf16, fuse biases ----------------
__global__ __launch_bounds__(256) void prep_weights_kernel(
    const float* __restrict__ wihf, const float* __restrict__ whhf,
    const float* __restrict__ bihf, const float* __restrict__ bhhf,
    const float* __restrict__ wihb, const float* __restrict__ whhb,
    const float* __restrict__ bihb, const float* __restrict__ bhhb,
    __bf16* __restrict__ Wih, __bf16* __restrict__ Whh, float* __restrict__ bias)
{
    const int i = blockIdx.x * 256 + threadIdx.x;
    if (i < G4 * EMBD) {
        Wih[i]             = to_bf16(wihf[i]);
        Whh[i]             = to_bf16(whhf[i]);
        Wih[G4 * EMBD + i] = to_bf16(wihb[i]);
        Whh[G4 * HD   + i] = to_bf16(whhb[i]);
    }
    if (i < G4)           bias[i]            = bihf[i] + bhhf[i];
    else if (i < 2 * G4)  bias[i]            = bihb[i - G4] + bhhb[i - G4];
}

// ---------------- K1: embedding gather -> bf16, time-major [s][b][e] ----------------
__global__ __launch_bounds__(256) void gather_kernel(
    const int* __restrict__ seqs, const float* __restrict__ emb, __bf16* __restrict__ X)
{
    const int tid = threadIdx.x;
    const int row = blockIdx.x * 16 + (tid >> 4);   // row = s*B + b
    const int s = row >> 7;
    const int b = row & 127;
    const int tok = seqs[(size_t)b * SB + s];
    const float* src = emb + (size_t)tok * EMBD + (tid & 15) * 8;
    float4 f0 = *(const float4*)(src);
    float4 f1 = *(const float4*)(src + 4);
    union { __bf16 h[8]; uint4 u; } pk;
    pk.h[0] = to_bf16(f0.x); pk.h[1] = to_bf16(f0.y);
    pk.h[2] = to_bf16(f0.z); pk.h[3] = to_bf16(f0.w);
    pk.h[4] = to_bf16(f1.x); pk.h[5] = to_bf16(f1.y);
    pk.h[6] = to_bf16(f1.z); pk.h[7] = to_bf16(f1.w);
    *(uint4*)(X + (size_t)row * EMBD + (tid & 15) * 8) = pk.u;
}

// ---------------- K2: input-gate GEMM  Gx[d][r][n] = X[r] @ Wih[d]^T + bias ----------------
// M = S*B = 65536, N = 512, K = 128. Block tile 128x64, 8 waves, 16 WMMA each.
// Tile staging via async global->LDS (no VGPR round trip; ASYNCcnt tracked).
__global__ __launch_bounds__(256) void input_gemm_kernel(
    const __bf16* __restrict__ X, const __bf16* __restrict__ Wih,
    const float* __restrict__ bias, float* __restrict__ Gx)
{
    __shared__ __align__(16) __bf16 ldsA[128 * EMBD];   // 32 KB
    __shared__ __align__(16) __bf16 ldsB[64  * EMBD];   // 16 KB
    const int mblk = blockIdx.x;     // 0..511
    const int nblk = blockIdx.y;     // 0..7
    const int d    = blockIdx.z;     // 0..1
    const int tid  = threadIdx.x;

    // A tile: 128 contiguous rows of X (one contiguous 32KB block) -> 8 async b128/thread
    const char* srcA = (const char*)(X + (size_t)mblk * 128 * EMBD);
#pragma unroll
    for (int i = 0; i < 8; ++i) {
        const size_t c = (size_t)(tid + 256 * i) * 16;
        async_b128((char*)ldsA + c, srcA + c);
    }
    // B tile: 64 contiguous rows of Wih[d] -> 4 async b128/thread
    const char* srcB = (const char*)(Wih + (size_t)d * G4 * EMBD + (size_t)nblk * 64 * EMBD);
#pragma unroll
    for (int i = 0; i < 4; ++i) {
        const size_t c = (size_t)(tid + 256 * i) * 16;
        async_b128((char*)ldsB + c, srcB + c);
    }
    if (mblk + 1 < 512)  // speculative prefetch of next M tile (global_prefetch_b8)
        __builtin_prefetch(X + (size_t)(mblk + 1) * 128 * EMBD + tid * 8, 0, 1);
    wait_async0();
    __syncthreads();

    const int w    = tid >> 5;
    const int lane = tid & 31;
    const int m0   = w * 16;
    v8f acc[4] = {};
#pragma unroll
    for (int kb = 0; kb < EMBD; kb += 32) {
        v16bf a = load_a_frag(ldsA, m0, kb, EMBD);
#pragma unroll
        for (int t = 0; t < 4; ++t) {
            v16bf b = load_b_frag(ldsB, t * 16, kb, EMBD);
            acc[t] = WMMA_BF16(a, b, acc[t]);
        }
    }
    // C layout: lane -> col n0+(lane&15); VGPR j -> row m0 + j + 8*(lane>=16)
    const int lm = (lane >> 4) << 3;
    const int ln = lane & 15;
    float* gbase = Gx + ((size_t)d * (SB * BB) + (size_t)mblk * 128) * G4;
#pragma unroll
    for (int t = 0; t < 4; ++t) {
        const int ng = nblk * 64 + t * 16 + ln;
        const float bv = bias[d * G4 + ng];
#pragma unroll
        for (int j = 0; j < 8; ++j)
            gbase[(size_t)(m0 + j + lm) * G4 + ng] = acc[t][j] + bv;
    }
}

// ---------------- K3: persistent bidirectional LSTM ----------------
// grid = (8 batch-chunks, 2 dirs), block = 256 (8 waves). Per step:
//   g = h @ Whh^T (+Gx), WMMA M=16 N=512 K=128 -> 32 tiles / 8 waves.
// LDS: Whh bf16 128KB (TDM-loaded) + h bf16 4KB + gates f32 32KB
//      + Gx double buffer 64KB (async-prefetched one step ahead) = 228KB (<320KB/WGP).
__global__ __launch_bounds__(256) void lstm_kernel(
    const __bf16* __restrict__ WhhAll, const float* __restrict__ Gx,
    float* __restrict__ Hout)
{
    extern __shared__ __align__(16) char smem[];
    __bf16* ldsW  = (__bf16*)smem;                                        // 512*128 bf16
    __bf16* ldsH  = (__bf16*)(smem + G4 * HD * 2);                        // 16*128 bf16
    float*  ldsG  = (float*) (smem + G4 * HD * 2 + 16 * HD * 2);          // 16*512 f32
    float*  ldsGx = (float*) (smem + G4 * HD * 2 + 16 * HD * 2 + 16 * G4 * 4); // 2x 16*512 f32

    const int bchunk = blockIdx.x;          // 0..7
    const int d      = blockIdx.y;          // 0..1
    const int tid    = threadIdx.x;
    const int bbase  = bchunk * 16;
    const float* gx0 = Gx + (size_t)d * (SB * BB) * G4;

#if __has_builtin(__builtin_amdgcn_tensor_load_to_lds)
    if (tid < 32)   // wave 0: one TDM descriptor DMAs all of Whh[d] (512x128 bf16)
        tdm_load_2d_bf16(ldsW, WhhAll + (size_t)d * G4 * HD, HD, G4);
#else
    {   // fallback: plain staged copy
        const uint4* srcW = (const uint4*)(WhhAll + (size_t)d * G4 * HD);
        uint4* dstW = (uint4*)ldsW;
#pragma unroll
        for (int i = 0; i < 32; ++i) dstW[tid + 256 * i] = srcW[tid + 256 * i];
    }
#endif
    // prefetch Gx slab for step 0 into buffer 0 (contiguous 32KB: 8 async b128/thread)
    {
        const int s0 = d ? (SB - 1) : 0;
        const char* src = (const char*)(gx0 + ((size_t)s0 * BB + bbase) * G4);
#pragma unroll
        for (int i = 0; i < 8; ++i) {
            const size_t c = (size_t)(tid + 256 * i) * 16;
            async_b128((char*)ldsGx + c, src + c);
        }
    }
    ((uint4*)ldsH)[tid] = make_uint4(0, 0, 0, 0);                   // h0 = 0

    float c[8];
#pragma unroll
    for (int q = 0; q < 8; ++q) c[q] = 0.0f;

    const int w    = tid >> 5;
    const int lane = tid & 31;
    const int n0   = w * 64;                 // wave's gate-column base
    const int lm   = (lane >> 4) << 3;
    const int ln   = lane & 15;
    const int pb   = tid >> 4;               // pointwise batch row 0..15
    const int pj   = (tid & 15) * 8;         // pointwise j base
    wait_async0();
    __syncthreads();

    for (int step = 0; step < SB; ++step) {
        const int s   = d ? (SB - 1 - step) : step;
        const int cur = step & 1;

        // issue async prefetch of next step's Gx slab into the other buffer
        if (step + 1 < SB) {
            const int s2 = d ? (SB - 2 - step) : (step + 1);
            const char* src = (const char*)(gx0 + ((size_t)s2 * BB + bbase) * G4);
            char* dst = (char*)(ldsGx + (1 - cur) * (16 * G4));
#pragma unroll
            for (int i = 0; i < 8; ++i) {
                const size_t cc = (size_t)(tid + 256 * i) * 16;
                async_b128(dst + cc, src + cc);
            }
        }

        v8f acc[4] = {};
#pragma unroll
        for (int kb = 0; kb < HD; kb += 32) {
            v16bf a = load_a_frag(ldsH, 0, kb, HD);
#pragma unroll
            for (int t = 0; t < 4; ++t) {
                v16bf b = load_b_frag(ldsW, n0 + t * 16, kb, HD);
                acc[t] = WMMA_BF16(a, b, acc[t]);
            }
        }
        // preact = recurrent + prefetched input part (biases already fused)
        const float* gxl = ldsGx + cur * (16 * G4);
#pragma unroll
        for (int t = 0; t < 4; ++t) {
            const int n = n0 + t * 16 + ln;
#pragma unroll
            for (int j = 0; j < 8; ++j) {
                const int m = j + lm;
                ldsG[m * G4 + n] = acc[t][j] + gxl[m * G4 + n];
            }
        }
        __syncthreads();

        // pointwise gates: this thread owns (pb, pj..pj+7); c lives in registers
        float* hrow = Hout + ((size_t)d * (SB * BB) + (size_t)s * BB + bbase + pb) * HD;
#pragma unroll
        for (int q = 0; q < 8; ++q) {
            const int j = pj + q;
            const float gi = ldsG[pb * G4 + j];
            const float gf = ldsG[pb * G4 + HD + j];
            const float gg = ldsG[pb * G4 + 2 * HD + j];
            const float go = ldsG[pb * G4 + 3 * HD + j];
            const float cn = sigmf(gf) * c[q] + sigmf(gi) * tanhf(gg);
            c[q] = cn;
            const float h = sigmf(go) * tanhf(cn);
            hrow[j] = h;
            ldsH[pb * HD + j] = to_bf16(h);
        }
        wait_async0();           // next-step Gx slab resident before anyone reads it
        __syncthreads();
    }
}

// ---------------- K4: emissions[b][s][l] = hcat . w_out[l] + b_out[l] ----------------
__global__ __launch_bounds__(256) void emissions_kernel(
    const float* __restrict__ Hout, const float* __restrict__ w_out,
    const float* __restrict__ b_out, float* __restrict__ em)
{
    const int idx = blockIdx.x * 256 + threadIdx.x;   // = s*B + b
    if (idx >= SB * BB) return;
    const int s = idx >> 7;
    const int b = idx & 127;
    const float* hf = Hout + ((size_t)0 * (SB * BB) + (size_t)s * BB + b) * HD;
    const float* hb = Hout + ((size_t)1 * (SB * BB) + (size_t)s * BB + b) * HD;
    float acc[LL];
#pragma unroll
    for (int l = 0; l < LL; ++l) acc[l] = b_out[l];
    for (int j = 0; j < HD; ++j) {
        const float hfv = hf[j], hbv = hb[j];
#pragma unroll
        for (int l = 0; l < LL; ++l)
            acc[l] += hfv * w_out[l * 256 + j] + hbv * w_out[l * 256 + HD + j];
    }
    float* out = em + ((size_t)b * SB + s) * LL;
#pragma unroll
    for (int l = 0; l < LL; ++l) out[l] = acc[l];
}

// ---------------- K5: CRF numerator + forward algorithm, one wave32 per batch row ----------------
__global__ __launch_bounds__(32) void crf_kernel(
    const int* __restrict__ seqs, const int* __restrict__ labels,
    const float* __restrict__ em, const float* __restrict__ start_t,
    const float* __restrict__ end_t, const float* __restrict__ trans,
    float* __restrict__ per_b)
{
    const int b    = blockIdx.x;
    const int lane = threadIdx.x;
    const int ls9  = (lane < LL) ? lane : 0;
    const float* e  = em + (size_t)b * SB * LL;
    const int* seq = seqs   + (size_t)b * SB;
    const int* lab = labels + (size_t)b * SB;

    // ---- numerator, strided over s then wave-reduced (uniform control flow) ----
    float numpart = 0.0f; int cnt = 0;
    for (int s = lane; s < SB; s += 32) {
        const int tok = seq[s];
        const int lv  = lab[s];
        if (s == 0) numpart += start_t[lv] + e[lv];
        else if (tok != 0) numpart += trans[lab[s - 1] * LL + lv] + e[s * LL + lv];
        if (tok != 0) cnt++;
    }
#pragma unroll
    for (int off = 16; off; off >>= 1) {
        numpart += __shfl_xor(numpart, off, 32);
        cnt     += __shfl_xor(cnt, off, 32);
    }
    float num = numpart + end_t[lab[cnt - 1]];

    // ---- forward algorithm: lane l holds score[l] ----
    float tcol[LL];
#pragma unroll
    for (int i = 0; i < LL; ++i) tcol[i] = trans[i * LL + ls9];
    float score = (lane < LL) ? (start_t[lane] + e[lane]) : -1e30f;

    for (int s = 1; s < SB; ++s) {
        const bool m = (seq[s] != 0);
        float tmp[LL];
        float mx = -1e30f;
#pragma unroll
        for (int i = 0; i < LL; ++i) {
            const float si = __shfl(score, i, 32);
            tmp[i] = si + tcol[i];
            mx = fmaxf(mx, tmp[i]);
        }
        float ssum = 0.0f;
#pragma unroll
        for (int i = 0; i < LL; ++i) ssum += __expf(tmp[i] - mx);
        const float nxt = mx + __logf(ssum) + e[s * LL + ls9];
        if (m && lane < LL) score = nxt;
    }

    // ---- denominator: logsumexp over lanes 0..8 ----
    float v = (lane < LL) ? (score + end_t[lane]) : -1e30f;
    float mx = v;
#pragma unroll
    for (int off = 16; off; off >>= 1) mx = fmaxf(mx, __shfl_xor(mx, off, 32));
    float ex = (lane < LL) ? __expf(v - mx) : 0.0f;
#pragma unroll
    for (int off = 16; off; off >>= 1) ex += __shfl_xor(ex, off, 32);
    if (lane == 0) per_b[b] = num - (mx + __logf(ex));
}

// ---------------- K6: final reduction -> d_out[0] = -sum(num - denom) ----------------
__global__ __launch_bounds__(128) void reduce_kernel(const float* __restrict__ per_b,
                                                     float* __restrict__ out)
{
    __shared__ float sh[128];
    const int t = threadIdx.x;
    sh[t] = per_b[t];
    __syncthreads();
#pragma unroll
    for (int off = 64; off; off >>= 1) {
        if (t < off) sh[t] += sh[t + off];
        __syncthreads();
    }
    if (t == 0) out[0] = -sh[0];
}

// ---------------- launcher ----------------
extern "C" void kernel_launch(void* const* d_in, const int* in_sizes, int n_in,
                              void* d_out, int out_size, void* d_ws, size_t ws_size,
                              hipStream_t stream) {
    const int*   seqs    = (const int*)  d_in[0];
    const int*   labels  = (const int*)  d_in[1];
    const float* emb     = (const float*)d_in[2];
    const float* w_ih_f  = (const float*)d_in[3];
    const float* w_hh_f  = (const float*)d_in[4];
    const float* b_ih_f  = (const float*)d_in[5];
    const float* b_hh_f  = (const float*)d_in[6];
    const float* w_ih_b  = (const float*)d_in[7];
    const float* w_hh_b  = (const float*)d_in[8];
    const float* b_ih_b  = (const float*)d_in[9];
    const float* b_hh_b  = (const float*)d_in[10];
    const float* w_out   = (const float*)d_in[11];
    const float* b_out   = (const float*)d_in[12];
    const float* start_t = (const float*)d_in[13];
    const float* end_t   = (const float*)d_in[14];
    const float* trans   = (const float*)d_in[15];

    char* ws = (char*)d_ws;
    size_t off = 0;
    __bf16* X    = (__bf16*)(ws + off); off += (size_t)SB * BB * EMBD * 2;          // 16 MB
    __bf16* Wih  = (__bf16*)(ws + off); off += (size_t)2 * G4 * EMBD * 2;           // 256 KB
    __bf16* Whh  = (__bf16*)(ws + off); off += (size_t)2 * G4 * HD * 2;             // 256 KB
    float*  bias = (float*) (ws + off); off += (size_t)2 * G4 * 4;                  // 4 KB
    float*  Gx   = (float*) (ws + off); off += (size_t)2 * SB * BB * G4 * 4;        // 256 MB
    float*  Hout = (float*) (ws + off); off += (size_t)2 * SB * BB * HD * 4;        // 64 MB
    float*  Em   = (float*) (ws + off); off += (size_t)BB * SB * LL * 4;            // 2.25 MB
    float*  Pb   = (float*) (ws + off);                                             // 512 B

    prep_weights_kernel<<<256, 256, 0, stream>>>(w_ih_f, w_hh_f, b_ih_f, b_hh_f,
                                                 w_ih_b, w_hh_b, b_ih_b, b_hh_b,
                                                 Wih, Whh, bias);
    gather_kernel<<<(SB * BB) / 16, 256, 0, stream>>>(seqs, emb, X);
    input_gemm_kernel<<<dim3(512, 8, 2), 256, 0, stream>>>(X, Wih, bias, Gx);

    const size_t lstm_lds = (size_t)G4 * HD * 2 + 16 * HD * 2
                          + 16 * G4 * 4 + 2 * 16 * G4 * 4;                          // 228 KB
    lstm_kernel<<<dim3(8, 2), 256, lstm_lds, stream>>>(Whh, Gx, Hout);

    emissions_kernel<<<(SB * BB) / 256, 256, 0, stream>>>(Hout, w_out, b_out, Em);
    crf_kernel<<<BB, 32, 0, stream>>>(seqs, labels, Em, start_t, end_t, trans, Pb);
    reduce_kernel<<<1, 128, 0, stream>>>(Pb, (float*)d_out);
}
